// Seq2Seq_61830349193962
// MI455X (gfx1250) — matrix operand order repaired and checked by reading, not compiled, not run on hardware
//
#include <hip/hip_runtime.h>

#define B_   512
#define S_   256
#define T_   48
#define DIN_ 16
#define E_   64
#define H_   128
#define G3_  384   // 3*H

typedef __attribute__((ext_vector_type(16))) _Float16 v16h;
typedef __attribute__((ext_vector_type(8)))  _Float16 v8h;
typedef __attribute__((ext_vector_type(8)))  float    v8f;
typedef _Float16 h16;

__device__ __forceinline__ float sigf(float x){ return 1.f/(1.f+__expf(-x)); }

// ---- WMMA fragment loaders (CDNA5 16-bit 16x32 A / 32x16 B layouts, wave32) ----
// A (16xK tile, row-major, ldk halfs): lane L: m=L&15, h=L>>4;
//   e=0..15 -> K = k0 + h*16 + e (contiguous)
__device__ __forceinline__ v16h load_A16(const h16* A, int ldk, int k0, int lane){
  const int m = lane & 15, hh = lane >> 4;
  const h16* p = A + (size_t)m*ldk + k0 + hh*16;
  v8h lo = *(const v8h*)(p);
  v8h hi = *(const v8h*)(p + 8);
  v16h a;
  #pragma unroll
  for(int e=0;e<8;++e){ a[e]=lo[e]; a[8+e]=hi[e]; }
  return a;
}
// B (Kx16 tile) from weight W stored (N x K) row-major f16: B(k,n)=W[n][k].
// lane L: n = n0 + (L&15), h = L>>4; elements e: K = k0 + h*16 + e (contiguous)
__device__ __forceinline__ v16h load_B16(const h16* W, int ldk, int n0, int k0, int lane){
  const int n = n0 + (lane & 15), hh = lane >> 4;
  const h16* p = W + (size_t)n*ldk + k0 + hh*16;
  return *(const v16h*)p;
}

// C = A(16xK) @ W(NxK)^T + bias -> LDS f32 C (16 x N), 8 waves cooperate.
__device__ __forceinline__ void gemm_lds(const h16* A, int K, const h16* __restrict__ W,
                                         const float* __restrict__ bias,
                                         float* C, int N, int lane, int wave){
  const int ktiles = K >> 5;
  for(int nt = wave; nt < (N >> 4); nt += 8){
    v8f c = {};
    for(int kt = 0; kt < ktiles; ++kt){
      v16h a = load_A16(A, K, kt*32, lane);
      v16h b = load_B16(W, K, nt*16, kt*32, lane);
      c = __builtin_amdgcn_wmma_f32_16x16x32_f16(false, a, false, b, (short)0, c, false, false);
    }
    const int n = nt*16 + (lane & 15), hh = lane >> 4;
    const float bv = bias[n];
    #pragma unroll
    for(int r=0;r<8;++r) C[(size_t)(r + 8*hh)*N + n] = c[r] + bv;
  }
}

// ---------------- prep kernels ----------------
__global__ void zero_kernel(float* p, int n){
  int i = blockIdx.x*blockDim.x + threadIdx.x;
  if(i < n) p[i] = 0.f;
}

__global__ void cvt_kernel(const float* __restrict__ src, int srcLd, int srcOff,
                           h16* __restrict__ dst, int n, int k, int kp){
  int idx = blockIdx.x*blockDim.x + threadIdx.x;
  if(idx >= n*kp) return;
  int r = idx/kp, c = idx%kp;
  dst[idx] = (c < k) ? (h16)src[(size_t)r*srcLd + srcOff + c] : (h16)0.f;
}

// fold embedding into layer0 input weight: fold = Wih0 @ emb_w  (384 x 16, K-padded to 32)
// foldb = Wih0 @ emb_b + bih0
__global__ void fold_kernel(const float* __restrict__ Wih, const float* __restrict__ embw,
                            const float* __restrict__ embb, const float* __restrict__ bih,
                            h16* __restrict__ fold16, float* __restrict__ foldb){
  int r = blockIdx.x*blockDim.x + threadIdx.x;
  if(r >= G3_) return;
  float bacc = bih[r];
  for(int e=0;e<E_;++e) bacc += Wih[(size_t)r*E_+e]*embb[e];
  foldb[r] = bacc;
  for(int c=0;c<32;++c){
    float a = 0.f;
    if(c < DIN_) for(int e=0;e<E_;++e) a += Wih[(size_t)r*E_+e]*embw[(size_t)e*DIN_+c];
    fold16[(size_t)r*32+c] = (h16)a;
  }
}

// ---------------- encoder: one time-step, both GRU layers ----------------
__global__ __launch_bounds__(256) void enc_step_kernel(int t,
    const float* __restrict__ past, const h16* __restrict__ fold16, const float* __restrict__ foldb,
    const h16* __restrict__ Whh0, const float* __restrict__ bhh0,
    const h16* __restrict__ Wih1, const float* __restrict__ bih1,
    const h16* __restrict__ Whh1, const float* __restrict__ bhh1,
    float* __restrict__ h0, float* __restrict__ h1, h16* __restrict__ enc_out16){
  __shared__ float sGI[16*G3_];
  __shared__ float sGH[16*G3_];
  __shared__ alignas(32) h16 sA1[16*H_];
  __shared__ alignas(32) h16 sA2[16*H_];
  __shared__ alignas(32) h16 sXP[16*32];
  const int tid = threadIdx.x, lane = tid & 31, wave = tid >> 5;
  const int m0 = blockIdx.x * 16;

  for(int idx = tid; idx < 16*32; idx += 256){
    int r = idx >> 5, c = idx & 31;
    sXP[idx] = (c < DIN_) ? (h16)past[((size_t)(m0+r)*S_ + t)*DIN_ + c] : (h16)0.f;
  }
  for(int idx = tid; idx < 16*H_; idx += 256){
    sA1[idx] = (h16)h0[(size_t)m0*H_ + idx];
    sA2[idx] = (h16)h1[(size_t)m0*H_ + idx];
  }
  __syncthreads();
  gemm_lds(sXP, 32,  fold16, foldb, sGI, G3_, lane, wave);   // gi0 (embedding folded in)
  gemm_lds(sA1, H_,  Whh0,   bhh0,  sGH, G3_, lane, wave);   // gh0
  __syncthreads();
  for(int idx = tid; idx < 16*H_; idx += 256){
    int m = idx >> 7, n = idx & 127;
    const float* gi = sGI + (size_t)m*G3_; const float* gh = sGH + (size_t)m*G3_;
    float r  = sigf(gi[n] + gh[n]);
    float z  = sigf(gi[H_+n] + gh[H_+n]);
    float nn = tanhf(gi[2*H_+n] + r*gh[2*H_+n]);
    float hold = h0[(size_t)m0*H_ + idx];
    float hp = (1.f - z)*nn + z*hold;
    h0[(size_t)m0*H_ + idx] = hp;
    sA1[idx] = (h16)hp;
  }
  __syncthreads();
  gemm_lds(sA1, H_, Wih1, bih1, sGI, G3_, lane, wave);       // gi1
  gemm_lds(sA2, H_, Whh1, bhh1, sGH, G3_, lane, wave);       // gh1
  __syncthreads();
  for(int idx = tid; idx < 16*H_; idx += 256){
    int m = idx >> 7, n = idx & 127;
    const float* gi = sGI + (size_t)m*G3_; const float* gh = sGH + (size_t)m*G3_;
    float r  = sigf(gi[n] + gh[n]);
    float z  = sigf(gi[H_+n] + gh[H_+n]);
    float nn = tanhf(gi[2*H_+n] + r*gh[2*H_+n]);
    float hold = h1[(size_t)m0*H_ + idx];
    float hp = (1.f - z)*nn + z*hold;
    h1[(size_t)m0*H_ + idx] = hp;
    enc_out16[((size_t)(m0+m)*S_ + t)*H_ + n] = (h16)hp;
  }
}

// ---------------- encoder FC: hidden = tanh(h1 @ fc^T + b) ----------------
__global__ __launch_bounds__(256) void fc_kernel(const float* __restrict__ h1,
    const h16* __restrict__ fc16, const float* __restrict__ fcb,
    float* __restrict__ hdf, h16* __restrict__ hd16){
  __shared__ alignas(32) h16 sA[16*H_];
  const int tid = threadIdx.x, lane = tid & 31, wave = tid >> 5;
  const int m0 = blockIdx.x*16;
  for(int idx = tid; idx < 16*H_; idx += 256) sA[idx] = (h16)h1[(size_t)m0*H_ + idx];
  __syncthreads();
  v8f c = {};
  #pragma unroll
  for(int kt=0;kt<4;++kt){
    v16h a = load_A16(sA, H_, kt*32, lane);
    v16h b = load_B16(fc16, H_, wave*16, kt*32, lane);
    c = __builtin_amdgcn_wmma_f32_16x16x32_f16(false,a,false,b,(short)0,c,false,false);
  }
  const int n = wave*16 + (lane&15), hh = lane>>4;
  const float bv = fcb[n];
  #pragma unroll
  for(int r=0;r<8;++r){
    float v = tanhf(c[r]+bv);
    size_t gi = (size_t)(m0 + r + 8*hh)*H_ + n;
    hdf[gi] = v; hd16[gi] = (h16)v;
  }
}

// ---------------- enc_proj = enc_out @ We^T + att_b (big parallel GEMM) ----------------
__global__ __launch_bounds__(256) void proj_kernel(const h16* __restrict__ enc_out16,
    const h16* __restrict__ We16, const float* __restrict__ attb, h16* __restrict__ enc_proj16){
  const int tid = threadIdx.x, lane = tid & 31, wave = tid >> 5;
  const size_t m0 = (size_t)blockIdx.x*16;
  const h16* A = enc_out16 + m0*H_;
  v8f c = {};
  #pragma unroll
  for(int kt=0;kt<4;++kt){
    v16h a = load_A16(A, H_, kt*32, lane);
    v16h b = load_B16(We16, H_, wave*16, kt*32, lane);
    c = __builtin_amdgcn_wmma_f32_16x16x32_f16(false,a,false,b,(short)0,c,false,false);
  }
  const int n = wave*16 + (lane&15), hh = lane>>4;
  const float bv = attb[n];
  #pragma unroll
  for(int r=0;r<8;++r) enc_proj16[(m0 + r + 8*hh)*H_ + n] = (h16)(c[r]+bv);
}

// ---------------- decoder pre: emb + h @ att_Wh^T ----------------
__global__ __launch_bounds__(256) void dec_pre_kernel(int t,
    const float* __restrict__ xs, const h16* __restrict__ demb16, const float* __restrict__ dembb,
    const h16* __restrict__ hd16, const h16* __restrict__ Wh16,
    float* __restrict__ emb32, h16* __restrict__ emb16, float* __restrict__ hWh){
  __shared__ alignas(32) h16 sXP[16*32];
  const int tid = threadIdx.x, lane = tid & 31, wave = tid >> 5;
  const int m0 = blockIdx.x*16;
  const int ts = (t==0) ? 0 : (t-1);
  for(int idx = tid; idx < 16*32; idx += 256){
    int r = idx >> 5, c = idx & 31;
    sXP[idx] = (c < DIN_) ? (h16)xs[((size_t)(m0+r)*T_ + ts)*DIN_ + c] : (h16)0.f;
  }
  __syncthreads();
  if(wave < 4){ // emb: N=64, K=32 (padded)
    v8f c = {};
    v16h a = load_A16(sXP, 32, 0, lane);
    v16h b = load_B16(demb16, 32, wave*16, 0, lane);
    c = __builtin_amdgcn_wmma_f32_16x16x32_f16(false,a,false,b,(short)0,c,false,false);
    const int n = wave*16 + (lane&15), hh = lane>>4;
    const float bv = dembb[n];
    #pragma unroll
    for(int r=0;r<8;++r){
      float v = c[r]+bv;
      size_t gi = (size_t)(m0 + r + 8*hh)*E_ + n;
      emb32[gi] = v; emb16[gi] = (h16)v;
    }
  }
  { // hWh: N=128, K=128
    const h16* A = hd16 + (size_t)m0*H_;
    v8f c = {};
    #pragma unroll
    for(int kt=0;kt<4;++kt){
      v16h a = load_A16(A, H_, kt*32, lane);
      v16h b = load_B16(Wh16, H_, wave*16, kt*32, lane);
      c = __builtin_amdgcn_wmma_f32_16x16x32_f16(false,a,false,b,(short)0,c,false,false);
    }
    const int n = wave*16 + (lane&15), hh = lane>>4;
    #pragma unroll
    for(int r=0;r<8;++r) hWh[(size_t)(m0 + r + 8*hh)*H_ + n] = c[r];
  }
}

// ---------------- decoder attention: scores, softmax, weighted sum ----------------
__global__ __launch_bounds__(256) void dec_att_kernel(
    const h16* __restrict__ enc_proj16, const h16* __restrict__ enc_out16,
    const float* __restrict__ hWh, const float* __restrict__ attv,
    float* __restrict__ w32, h16* __restrict__ w16v){
  __shared__ float sQ[H_], sV[H_], sSC[S_], sRED[256];
  const int tid = threadIdx.x, lane = tid & 31, wave = tid >> 5;
  const size_t b = blockIdx.x;
  if(tid < H_){ sQ[tid] = hWh[b*H_+tid]; sV[tid] = attv[tid]; }
  __syncthreads();
  const h16* EP = enc_proj16 + b*S_*H_;
  for(int s = wave*32; s < wave*32+32; ++s){
    float p = 0.f;
    const h16* row = EP + (size_t)s*H_;
    for(int h = lane; h < H_; h += 32) p += tanhf((float)row[h] + sQ[h]) * sV[h];
    #pragma unroll
    for(int off=16; off; off>>=1) p += __shfl_xor(p, off, 32);
    if(lane == 0) sSC[s] = p;
  }
  __syncthreads();
  float sc = sSC[tid];
  sRED[tid] = sc; __syncthreads();
  for(int off=128; off; off>>=1){ if(tid<off) sRED[tid]=fmaxf(sRED[tid], sRED[tid+off]); __syncthreads(); }
  float mx = sRED[0]; __syncthreads();
  float ex = __expf(sc - mx);
  sRED[tid] = ex; __syncthreads();
  for(int off=128; off; off>>=1){ if(tid<off) sRED[tid]+=sRED[tid+off]; __syncthreads(); }
  float inv = 1.f / sRED[0]; __syncthreads();
  sSC[tid] = ex * inv;
  __syncthreads();
  const h16* EO = enc_out16 + b*S_*H_;
  {
    int h = tid & 127, half = tid >> 7;
    float acc = 0.f;
    for(int s = half*128; s < half*128+128; ++s) acc += sSC[s]*(float)EO[(size_t)s*H_ + h];
    sRED[tid] = acc;
  }
  __syncthreads();
  if(tid < H_){
    float v = sRED[tid] + sRED[tid+128];
    w32[b*H_+tid] = v; w16v[b*H_+tid] = (h16)v;
  }
}

// ---------------- decoder GRU cell + output row ----------------
__global__ __launch_bounds__(256) void dec_cell_kernel(int t,
    const h16* __restrict__ emb16, const float* __restrict__ emb32,
    const h16* __restrict__ w16v,  const float* __restrict__ w32v,
    const h16* __restrict__ dWih16, const float* __restrict__ dbih,
    const h16* __restrict__ dWhh16, const float* __restrict__ dbhh,
    const float* __restrict__ outw, const float* __restrict__ outb,
    float* __restrict__ hdf, h16* __restrict__ hd16, float* __restrict__ out){
  __shared__ float sGI[16*G3_];
  __shared__ float sGH[16*G3_];
  __shared__ alignas(32) unsigned char sR0[16*192*2 + 16*H_*2];   // cat(f16)+h(f16); reused as h2(f32)
  h16*   sCAT = (h16*)sR0;                 // 16 x 192
  h16*   sH   = (h16*)(sR0 + 16*192*2);    // 16 x 128
  float* sHN  = (float*)sR0;               // 16 x 128 f32 (alias, used after GEMMs done)
  const int tid = threadIdx.x, lane = tid & 31, wave = tid >> 5;
  const int m0 = blockIdx.x*16;
  for(int idx = tid; idx < 16*192; idx += 256){
    int m = idx/192, c = idx%192;
    sCAT[idx] = (c < E_) ? emb16[(size_t)(m0+m)*E_ + c] : w16v[(size_t)(m0+m)*H_ + (c-E_)];
  }
  for(int idx = tid; idx < 16*H_; idx += 256) sH[idx] = hd16[(size_t)m0*H_ + idx];
  __syncthreads();
  gemm_lds(sCAT, 192, dWih16, dbih, sGI, G3_, lane, wave);
  gemm_lds(sH,   H_,  dWhh16, dbhh, sGH, G3_, lane, wave);
  __syncthreads();
  for(int idx = tid; idx < 16*H_; idx += 256){
    int m = idx >> 7, n = idx & 127;
    const float* gi = sGI + (size_t)m*G3_; const float* gh = sGH + (size_t)m*G3_;
    float r  = sigf(gi[n] + gh[n]);
    float z  = sigf(gi[H_+n] + gh[H_+n]);
    float nn = tanhf(gi[2*H_+n] + r*gh[2*H_+n]);
    float hold = hdf[(size_t)m0*H_ + idx];
    float hp = (1.f - z)*nn + z*hold;
    hdf[(size_t)m0*H_ + idx]  = hp;
    hd16[(size_t)m0*H_ + idx] = (h16)hp;
    sHN[idx] = hp;
  }
  __syncthreads();
  if(tid < 16){
    int m = tid;
    float acc = outb[0];
    for(int h=0;h<H_;++h) acc += sHN[m*H_+h]*outw[h];
    for(int h=0;h<H_;++h) acc += w32v[(size_t)(m0+m)*H_+h]*outw[H_+h];
    for(int c=0;c<E_;++c) acc += emb32[(size_t)(m0+m)*E_+c]*outw[2*H_+c];
    out[(size_t)(m0+m)*T_ + t] = sigf(acc);
  }
}

// ---------------- host ----------------
extern "C" void kernel_launch(void* const* d_in, const int* in_sizes, int n_in,
                              void* d_out, int out_size, void* d_ws, size_t ws_size,
                              hipStream_t stream){
  (void)in_sizes; (void)n_in; (void)out_size; (void)ws_size;
  const float* past  = (const float*)d_in[0];
  const float* xs    = (const float*)d_in[1];
  const float* eembw = (const float*)d_in[2];
  const float* eembb = (const float*)d_in[3];
  const float* Wih0  = (const float*)d_in[4];
  const float* Whh0  = (const float*)d_in[5];
  const float* bih0  = (const float*)d_in[6];
  const float* bhh0  = (const float*)d_in[7];
  const float* Wih1  = (const float*)d_in[8];
  const float* Whh1  = (const float*)d_in[9];
  const float* bih1  = (const float*)d_in[10];
  const float* bhh1  = (const float*)d_in[11];
  const float* fcw   = (const float*)d_in[12];
  const float* fcb   = (const float*)d_in[13];
  const float* attw  = (const float*)d_in[14];
  const float* attb  = (const float*)d_in[15];
  const float* attv  = (const float*)d_in[16];
  const float* dembw = (const float*)d_in[17];
  const float* dembb = (const float*)d_in[18];
  const float* dWih  = (const float*)d_in[19];
  const float* dWhh  = (const float*)d_in[20];
  const float* dbih  = (const float*)d_in[21];
  const float* dbhh  = (const float*)d_in[22];
  const float* outw  = (const float*)d_in[23];
  const float* outb  = (const float*)d_in[24];
  float* out = (float*)d_out;

  unsigned char* ws = (unsigned char*)d_ws;
  size_t off = 0;
  auto take = [&](size_t bytes)->void*{
    void* p = ws + off; off = (off + bytes + 255) & ~(size_t)255; return p;
  };
  float* h0f        = (float*)take((size_t)B_*H_*4);
  float* h1f        = (float*)take((size_t)B_*H_*4);
  float* hdf        = (float*)take((size_t)B_*H_*4);
  h16*   hd16       = (h16*)  take((size_t)B_*H_*2);
  h16*   enc_out16  = (h16*)  take((size_t)B_*S_*H_*2);
  h16*   enc_proj16 = (h16*)  take((size_t)B_*S_*H_*2);
  float* emb32      = (float*)take((size_t)B_*E_*4);
  h16*   emb16      = (h16*)  take((size_t)B_*E_*2);
  float* w32        = (float*)take((size_t)B_*H_*4);
  h16*   w16        = (h16*)  take((size_t)B_*H_*2);
  float* hWh        = (float*)take((size_t)B_*H_*4);
  h16*   fold16     = (h16*)  take((size_t)G3_*32*2);
  float* foldb      = (float*)take((size_t)G3_*4);
  h16*   Whh0_16    = (h16*)  take((size_t)G3_*H_*2);
  h16*   Wih1_16    = (h16*)  take((size_t)G3_*H_*2);
  h16*   Whh1_16    = (h16*)  take((size_t)G3_*H_*2);
  h16*   fc16       = (h16*)  take((size_t)H_*H_*2);
  h16*   We16       = (h16*)  take((size_t)H_*H_*2);
  h16*   Wh16       = (h16*)  take((size_t)H_*H_*2);
  h16*   demb16     = (h16*)  take((size_t)E_*32*2);
  h16*   dWih16     = (h16*)  take((size_t)G3_*192*2);
  h16*   dWhh16     = (h16*)  take((size_t)G3_*H_*2);

  auto cgrid = [](int n){ return (n + 255)/256; };

  // prep: weight conversion + embedding fold + state init
  fold_kernel<<<cgrid(G3_), 256, 0, stream>>>(Wih0, eembw, eembb, bih0, fold16, foldb);
  cvt_kernel<<<cgrid(G3_*H_),  256, 0, stream>>>(Whh0, H_, 0,  Whh0_16, G3_, H_, H_);
  cvt_kernel<<<cgrid(G3_*H_),  256, 0, stream>>>(Wih1, H_, 0,  Wih1_16, G3_, H_, H_);
  cvt_kernel<<<cgrid(G3_*H_),  256, 0, stream>>>(Whh1, H_, 0,  Whh1_16, G3_, H_, H_);
  cvt_kernel<<<cgrid(H_*H_),   256, 0, stream>>>(fcw,  H_, 0,  fc16,  H_, H_, H_);
  cvt_kernel<<<cgrid(H_*H_),   256, 0, stream>>>(attw, 2*H_, 0,   We16, H_, H_, H_);
  cvt_kernel<<<cgrid(H_*H_),   256, 0, stream>>>(attw, 2*H_, H_,  Wh16, H_, H_, H_);
  cvt_kernel<<<cgrid(E_*32),   256, 0, stream>>>(dembw, DIN_, 0, demb16, E_, DIN_, 32);
  cvt_kernel<<<cgrid(G3_*192), 256, 0, stream>>>(dWih, 192, 0, dWih16, G3_, 192, 192);
  cvt_kernel<<<cgrid(G3_*H_),  256, 0, stream>>>(dWhh, H_, 0,  dWhh16, G3_, H_, H_);
  zero_kernel<<<cgrid(B_*H_), 256, 0, stream>>>(h0f, B_*H_);
  zero_kernel<<<cgrid(B_*H_), 256, 0, stream>>>(h1f, B_*H_);

  // encoder recurrence (serial over S)
  for(int t = 0; t < S_; ++t){
    enc_step_kernel<<<B_/16, 256, 0, stream>>>(t, past, fold16, foldb,
        Whh0_16, bhh0, Wih1_16, bih1, Whh1_16, bhh1, h0f, h1f, enc_out16);
  }
  fc_kernel<<<B_/16, 256, 0, stream>>>(h1f, fc16, fcb, hdf, hd16);
  proj_kernel<<<(B_*S_)/16, 256, 0, stream>>>(enc_out16, We16, attb, enc_proj16);

  // decoder recurrence (serial over T)
  for(int t = 0; t < T_; ++t){
    dec_pre_kernel<<<B_/16, 256, 0, stream>>>(t, xs, demb16, dembb, hd16, Wh16,
                                              emb32, emb16, hWh);
    dec_att_kernel<<<B_, 256, 0, stream>>>(enc_proj16, enc_out16, hWh, attv, w32, w16);
    dec_cell_kernel<<<B_/16, 256, 0, stream>>>(t, emb16, emb32, w16, w32,
        dWih16, dbih, dWhh16, dbhh, outw, outb, hdf, hd16, out);
  }
}
